// CRISPROModel_4827543241084
// MI455X (gfx1250) — matrix operand order repaired and checked by reading, not compiled, not run on hardware
//
#include <hip/hip_runtime.h>
#include <hip/hip_bf16.h>
#include <math.h>

// ---------------------------------------------------------------------------
// Model dims (match reference)
// ---------------------------------------------------------------------------
#define Bn        16
#define Ln        1024
#define Mtok      (Bn * Ln)          // 16384 tokens
#define D_MODEL   256
#define N_LAYERS  4
#define D_STATE   128
#define HEADDIM   64
#define D_INNER   512
#define NHEADS    8
#define D_CONV    4
#define D_XBC     768                // D_INNER + 2*D_STATE
#define D_INPROJ  1288               // 2*D_INNER + 2*D_STATE + NHEADS
#define NPAD_ST   1296               // zx row stride (D_INPROJ -> mult of 16)
#define NPAD_WIN  1408               // in_proj Wt rows (mult of 128 for 8-wave blocks)

// ---------------------------------------------------------------------------
// Optional gfx1250 async global->LDS path (guarded; falls back to LDS stores)
// Probe result (round 2): builtin exists; param0 is v4i* (16B int vector ptr).
// ---------------------------------------------------------------------------
#if defined(__has_builtin)
#if __has_builtin(__builtin_amdgcn_global_load_async_to_lds_b128) && \
    __has_builtin(__builtin_amdgcn_s_wait_asynccnt)
#define HAVE_ASYNC 1
#endif
#endif
#ifndef HAVE_ASYNC
#define HAVE_ASYNC 0
#endif

#if HAVE_ASYNC
typedef int v4i_async __attribute__((vector_size(16)));
__device__ __forceinline__ void async_g2l_b128(const void* g, void* l) {
    __builtin_amdgcn_global_load_async_to_lds_b128(
        (v4i_async*)g, (v4i_async*)l, 0, 0);
}
__device__ __forceinline__ void async_wait0() {
    __builtin_amdgcn_s_wait_asynccnt(0);
}
#endif

// ---------------------------------------------------------------------------
// WMMA types (CDNA5 / gfx1250, wave32)
// ---------------------------------------------------------------------------
typedef __bf16 bf16_t;
typedef bf16_t v16bf __attribute__((ext_vector_type(16)));
typedef float  v8f   __attribute__((ext_vector_type(8)));

union FragAB { unsigned u[8]; v16bf v; };

__device__ __forceinline__ unsigned short f32_to_bf16_rne(float f) {
    unsigned u = __float_as_uint(f);
    unsigned r = u + 0x7FFFu + ((u >> 16) & 1u);
    return (unsigned short)(r >> 16);
}
__device__ __forceinline__ float sigmoidf_(float x) { return 1.0f / (1.0f + expf(-x)); }
__device__ __forceinline__ float siluf_(float x)    { return x / (1.0f + expf(-x)); }

__device__ __forceinline__ float block_reduce_sum256(float v, float* red) {
    int t = threadIdx.x;
    red[t] = v; __syncthreads();
    for (int s = 128; s > 0; s >>= 1) {
        if (t < s) red[t] += red[t + s];
        __syncthreads();
    }
    float r = red[0]; __syncthreads();
    return r;
}

// ---------------------------------------------------------------------------
// Kernel 1: embedding + epi projection + gating  -> h (Mtok x 256)
// ---------------------------------------------------------------------------
__global__ void k_embed_gate(const int* __restrict__ xtok,
                             const float* __restrict__ epi,      // Mtok x 33
                             const float* __restrict__ emb,      // 20 x 256
                             const float* __restrict__ epi_w,    // 33 x 256
                             const float* __restrict__ epi_b,    // 256
                             const float* __restrict__ gate_w,   // 512 x 256
                             const float* __restrict__ gate_b,   // 256
                             float* __restrict__ hout)           // Mtok x 256
{
    __shared__ float shh[D_MODEL];
    __shared__ float she[D_MODEL];
    int row = blockIdx.x;
    int c   = threadIdx.x;
    int tok = xtok[row];
    float hv = emb[tok * D_MODEL + c];
    const float* ep = epi + (size_t)row * 33;
    float e = epi_b[c];
    #pragma unroll
    for (int j = 0; j < 33; ++j) e += ep[j] * epi_w[j * D_MODEL + c];
    shh[c] = hv; she[c] = e;
    __syncthreads();
    float g = gate_b[c];
    for (int j = 0; j < D_MODEL; ++j) g += shh[j] * gate_w[j * D_MODEL + c];
    for (int j = 0; j < D_MODEL; ++j) g += she[j] * gate_w[(D_MODEL + j) * D_MODEL + c];
    g = sigmoidf_(g);
    hout[(size_t)row * D_MODEL + c] = hv + g * e;
}

// ---------------------------------------------------------------------------
// Kernel 2: RMSNorm over 256 dims -> bf16, plus residual copy into accumulator
// ---------------------------------------------------------------------------
__global__ void k_rmsnorm_copy(const float* __restrict__ h,
                               const float* __restrict__ w,
                               unsigned short* __restrict__ hnb,  // Mtok x 256 bf16
                               float* __restrict__ acc)           // Mtok x 256
{
    __shared__ float red[256];
    int row = blockIdx.x, t = threadIdx.x;
    float v = h[(size_t)row * D_MODEL + t];
    float ss = block_reduce_sum256(v * v, red);
    float r = rsqrtf(ss / (float)D_MODEL + 1e-5f);
    float o = v * r * w[t];
    hnb[(size_t)row * D_MODEL + t] = f32_to_bf16_rne(o);
    acc[(size_t)row * D_MODEL + t] = v;
}

// ---------------------------------------------------------------------------
// Kernel 3: weight convert + transpose:  W (K x Nsrc, f32) -> Wt (Npad x K, bf16)
// ---------------------------------------------------------------------------
__global__ void k_convert_wt(const float* __restrict__ W,
                             unsigned short* __restrict__ Wt,
                             int K, int Nsrc, int Npad)
{
    int i = blockIdx.x * blockDim.x + threadIdx.x;
    if (i >= Npad * K) return;
    int n = i / K, k = i - n * K;
    float v = (n < Nsrc) ? W[(size_t)k * Nsrc + n] : 0.0f;
    Wt[i] = f32_to_bf16_rne(v);
}

// ---------------------------------------------------------------------------
// Kernel 4: WMMA bf16 GEMM with LDS-staged A slab shared by 8 waves.
// Block = 256 threads: computes a 16(M) x 128(N) C tile; wave w -> n-tile w.
// grid = dim3(Nw/128, M/16).  Wt rows (Nw) must be a multiple of 128.
// C[M x ldc] (+)= A[M x K] * Wt[Nw x K]^T  (stores guarded to col < Nstore)
// ---------------------------------------------------------------------------
__global__ void k_gemm_wmma_bf16(const unsigned short* __restrict__ A,
                                 const unsigned short* __restrict__ Wt,
                                 float* __restrict__ C,
                                 int Nstore, int K, int ldc, int accumulate)
{
    __shared__ __align__(16) unsigned short sA[16 * 512];   // 16 x K (K <= 512)
    const int tid  = threadIdx.x;
    const int wave = tid >> 5;
    const int lane = tid & 31;
    const int mt   = blockIdx.y;
    const int nt   = blockIdx.x * 8 + wave;

    // ---- stage A slab (16 rows x K bf16) into LDS, 16B per transfer ----
    {
        const int slabBytes = 16 * K * 2;
        const char* Ag = (const char*)(A + (size_t)mt * 16 * K);
        char*       Al = (char*)sA;
        for (int o = tid * 16; o < slabBytes; o += 256 * 16) {
#if HAVE_ASYNC
            async_g2l_b128(Ag + o, Al + o);
#else
            *(uint4*)(Al + o) = *(const uint4*)(Ag + o);
#endif
        }
#if HAVE_ASYNC
        async_wait0();
#endif
        __syncthreads();
    }

    const int half = lane >> 4;
    const int r    = lane & 15;
    const unsigned* ArowL = (const unsigned*)sA + r * (K >> 1);
    const unsigned* Brow  = (const unsigned*)(Wt + (size_t)(nt * 16 + r) * K);

    v8f acc = {};
    for (int kk = 0; kk < K; kk += 32) {
        int kw = kk >> 1;                            // dword index into the row
        __builtin_prefetch(Brow + kw + 16, 0, 1);    // next B K-slab
        FragAB a, b;
        #pragma unroll
        for (int j = 0; j < 8; ++j) {
            // 16-bit A 16x32 frag: VGPR j<4 -> K=8*half+2j ; j>=4 -> K=16+8*half+2(j-4)
            int ka = (j < 4) ? (4 * half + j) : (8 + 4 * half + (j - 4));
            a.u[j] = ArowL[kw + ka];
            // B 32x16 (stored N x K): lane col, same K order as A
            b.u[j] = Brow[kw + ka];
        }
        acc = __builtin_amdgcn_wmma_f32_16x16x32_bf16(
                  false, a.v, false, b.v, (short)0, acc, false, false);
    }

    // C f32 16x16 layout: VGPR i -> row (8*half + i), col = lane&15
    int col = nt * 16 + r;
    #pragma unroll
    for (int i = 0; i < 8; ++i) {
        int rowm = mt * 16 + half * 8 + i;
        if (col < Nstore) {
            size_t off = (size_t)rowm * ldc + col;
            C[off] = accumulate ? (C[off] + acc[i]) : acc[i];
        }
    }
}

// ---------------------------------------------------------------------------
// Kernel 5: direction-aware depthwise causal conv (D_CONV=4) + bias + SiLU
// ---------------------------------------------------------------------------
__global__ void k_conv_silu(const float* __restrict__ zx,      // Mtok x NPAD_ST
                            const float* __restrict__ conv_w,  // 768 x 4
                            const float* __restrict__ conv_b,  // 768
                            float* __restrict__ xbc,           // Mtok x 768
                            int dir)
{
    int idx = blockIdx.x * blockDim.x + threadIdx.x;
    if (idx >= Mtok * D_XBC) return;
    int c  = idx % D_XBC;
    int tk = (idx / D_XBC) % Ln;
    int b  = idx / (D_XBC * Ln);
    float a = conv_b[c];
    #pragma unroll
    for (int k = 0; k < D_CONV; ++k) {
        int tt = (dir == 0) ? (tk + k - (D_CONV - 1)) : (tk + (D_CONV - 1) - k);
        if (tt >= 0 && tt < Ln)
            a += zx[(size_t)(b * Ln + tt) * NPAD_ST + D_INNER + c] * conv_w[c * D_CONV + k];
    }
    xbc[(size_t)(b * Ln + tk) * D_XBC + c] = siluf_(a);
}

// ---------------------------------------------------------------------------
// Kernel 6: selective-state scan. One block per (batch, head).
// State 64x128 in registers (32/thread); B_t,C_t staged (async) into LDS.
// ---------------------------------------------------------------------------
__global__ void k_scan(const float* __restrict__ zx,       // dt at col 1280+h
                       const float* __restrict__ xbc,      // Mtok x 768
                       const float* __restrict__ dt_bias,  // 8
                       const float* __restrict__ A_log,    // 8
                       const float* __restrict__ Dp,       // 8
                       float* __restrict__ y,              // Mtok x 512
                       int dir)
{
    __shared__ __align__(16) float sBC[2 * D_STATE];   // [0,128)=B_t, [128,256)=C_t
    __shared__ float sS[2];                            // dt, dA
    int bh = blockIdx.x;
    int b  = bh / NHEADS;
    int h  = bh - b * NHEADS;
    int t  = threadIdx.x;
    int p   = t >> 2;
    int sub = t & 3;
    int nb  = sub * 32;

    float s[32];
    #pragma unroll
    for (int i = 0; i < 32; ++i) s[i] = 0.0f;

    float negA = -expf(A_log[h]);
    float Dh   = Dp[h];

    for (int step = 0; step < Ln; ++step) {
        int tok = (dir == 0) ? step : (Ln - 1 - step);
        size_t row  = (size_t)b * Ln + tok;
        size_t base = row * D_XBC;
        if (t < 64) {      // stage 256 floats (B||C) in 16B chunks
#if HAVE_ASYNC
            async_g2l_b128(xbc + base + D_INNER + t * 4, sBC + t * 4);
#else
            *(float4*)(sBC + t * 4) = *(const float4*)(xbc + base + D_INNER + t * 4);
#endif
        }
#if HAVE_ASYNC
        async_wait0();
#endif
        if (t == 0) {
            float a = zx[row * NPAD_ST + (D_INNER + D_XBC) + h] + dt_bias[h];
            float dt = (a > 20.0f) ? a : log1pf(expf(a));
            sS[0] = dt;
            sS[1] = expf(negA * dt);
        }
        __syncthreads();
        float xv  = xbc[base + h * HEADDIM + p];
        float dbx = sS[0] * xv;
        float dA  = sS[1];
        float part = 0.0f;
        #pragma unroll
        for (int i = 0; i < 32; ++i) {
            float sv = dA * s[i] + dbx * sBC[nb + i];
            s[i] = sv;
            part += sv * sBC[D_STATE + nb + i];
        }
        part += __shfl_xor(part, 1, 32);
        part += __shfl_xor(part, 2, 32);
        if (sub == 0)
            y[row * D_INNER + h * HEADDIM + p] = part + Dh * xv;
        __syncthreads();
    }
}

// ---------------------------------------------------------------------------
// Kernel 7: u = rmsnorm(y * silu(z)) * norm_w -> bf16
// ---------------------------------------------------------------------------
__global__ void k_gatenorm(const float* __restrict__ zx,      // z in cols [0,512)
                           const float* __restrict__ y,       // Mtok x 512
                           const float* __restrict__ norm_w,  // 512
                           unsigned short* __restrict__ ub)   // Mtok x 512 bf16
{
    __shared__ float red[256];
    int row = blockIdx.x, t = threadIdx.x;
    size_t zb = (size_t)row * NPAD_ST;
    size_t yb = (size_t)row * D_INNER;
    float z0 = zx[zb + t],        z1 = zx[zb + t + 256];
    float v0 = y[yb + t]       * siluf_(z0);
    float v1 = y[yb + t + 256] * siluf_(z1);
    float ss = block_reduce_sum256(v0 * v0 + v1 * v1, red);
    float r = rsqrtf(ss / (float)D_INNER + 1e-5f);
    ub[yb + t]       = f32_to_bf16_rne(v0 * r * norm_w[t]);
    ub[yb + t + 256] = f32_to_bf16_rne(v1 * r * norm_w[t + 256]);
}

// ---------------------------------------------------------------------------
// Kernel 8: zero helper
// ---------------------------------------------------------------------------
__global__ void k_zero(float* __restrict__ p, int n) {
    int i = blockIdx.x * blockDim.x + threadIdx.x;
    if (i < n) p[i] = 0.0f;
}

// ---------------------------------------------------------------------------
// Kernel 9: final layernorm + mean pool over L (atomic accumulation)
// ---------------------------------------------------------------------------
__global__ void k_finalln_pool(const float* __restrict__ h,
                               const float* __restrict__ w,
                               const float* __restrict__ bws,
                               float* __restrict__ pooled)    // Bn x 256
{
    __shared__ float red[256];
    int row = blockIdx.x, t = threadIdx.x;
    int b = row / Ln;
    float v = h[(size_t)row * D_MODEL + t];
    float mu = block_reduce_sum256(v, red) / (float)D_MODEL;
    float c  = v - mu;
    float var = block_reduce_sum256(c * c, red) / (float)D_MODEL;
    float o = c * rsqrtf(var + 1e-5f) * w[t] + bws[t];
    atomicAdd(&pooled[b * D_MODEL + t], o * (1.0f / (float)Ln));
}

// ---------------------------------------------------------------------------
// Kernel 10: biophysics gate + cls/reg heads
// ---------------------------------------------------------------------------
__global__ void k_heads(const float* __restrict__ pooled,
                        const float* __restrict__ bio,      // Bn x 2
                        const float* __restrict__ phys_w,   // 2 x 256
                        const float* __restrict__ phys_b,   // 256
                        const float* __restrict__ cls_w,    // 256
                        const float* __restrict__ cls_b,    // 1
                        const float* __restrict__ reg_w,    // 256
                        const float* __restrict__ reg_b,    // 1
                        float* __restrict__ out)            // [cls(16), reg(16)]
{
    __shared__ float red[256];
    int b = blockIdx.x, t = threadIdx.x;
    float g2 = sigmoidf_(bio[b * 2 + 0] * phys_w[t] +
                         bio[b * 2 + 1] * phys_w[D_MODEL + t] + phys_b[t]);
    float pc = pooled[b * D_MODEL + t] * g2;
    float cls = block_reduce_sum256(pc * cls_w[t], red);
    float reg = block_reduce_sum256(pc * reg_w[t], red);
    if (t == 0) {
        out[b]      = cls + cls_b[0];
        out[Bn + b] = reg + reg_b[0];
    }
}

// ---------------------------------------------------------------------------
// Host launcher
// d_in flattening assumption: depth-first, dict-insertion order:
//   0:x 1:epigenetics 2:biophysics
//   3:emb 4:epi_w 5:epi_b 6:gate_w 7:gate_b 8:final_ln_w 9:final_ln_b
//   10:phys_w 11:phys_b 12:cls_w 13:cls_b 14:reg_w 15:reg_b
//   16+17*l: { ln_w, fwd:{in_proj,conv_w,conv_b,dt_bias,A_log,D,norm_w,out_proj},
//              bwd:{same 8} }
// ---------------------------------------------------------------------------
extern "C" void kernel_launch(void* const* d_in, const int* in_sizes, int n_in,
                              void* d_out, int out_size, void* d_ws, size_t ws_size,
                              hipStream_t stream) {
    (void)in_sizes; (void)n_in; (void)out_size; (void)ws_size;
    const int*   xtok = (const int*)d_in[0];
    const float* epi  = (const float*)d_in[1];
    const float* bio  = (const float*)d_in[2];
    #define FP(i) ((const float*)d_in[(i)])

    // ---- workspace carve-out -------------------------------------------
    char* ws = (char*)d_ws;
    size_t off = 0;
    auto carve = [&](size_t bytes) -> char* {
        char* p = ws + off;
        off += (bytes + 255) & ~(size_t)255;
        return p;
    };
    float*          h0     = (float*)carve((size_t)Mtok * D_MODEL * 4);
    float*          h1     = (float*)carve((size_t)Mtok * D_MODEL * 4);
    unsigned short* hnb    = (unsigned short*)carve((size_t)Mtok * D_MODEL * 2);
    float*          zx     = (float*)carve((size_t)Mtok * NPAD_ST * 4);
    float*          xbc    = (float*)carve((size_t)Mtok * D_XBC * 4);
    float*          ybuf   = (float*)carve((size_t)Mtok * D_INNER * 4);
    unsigned short* ub     = (unsigned short*)carve((size_t)Mtok * D_INNER * 2);
    unsigned short* wt_in  = (unsigned short*)carve((size_t)NPAD_WIN * D_MODEL * 2);
    unsigned short* wt_out = (unsigned short*)carve((size_t)D_MODEL * D_INNER * 2);
    float*          pooled = (float*)carve((size_t)Bn * D_MODEL * 4);

    // ---- front end ------------------------------------------------------
    k_embed_gate<<<Mtok, 256, 0, stream>>>(xtok, epi, FP(3), FP(4), FP(5),
                                           FP(6), FP(7), h0);

    float* hcur = h0;
    float* hacc = h1;
    const dim3 grid_in(NPAD_WIN / 128, Mtok / 16);   // (11, 1024)
    const dim3 grid_out(2, Mtok / 16);               // 256 cols -> 2 blocks of 128

    for (int l = 0; l < N_LAYERS; ++l) {
        int lb = 16 + 17 * l;
        k_rmsnorm_copy<<<Mtok, 256, 0, stream>>>(hcur, FP(lb + 0), hnb, hacc);
        for (int d = 0; d < 2; ++d) {
            int db = lb + 1 + d * 8;
            const float* in_proj  = FP(db + 0);
            const float* conv_w   = FP(db + 1);
            const float* conv_b   = FP(db + 2);
            const float* dt_bias  = FP(db + 3);
            const float* A_log    = FP(db + 4);
            const float* Dparam   = FP(db + 5);
            const float* norm_w   = FP(db + 6);
            const float* out_proj = FP(db + 7);

            k_convert_wt<<<(NPAD_WIN * D_MODEL + 255) / 256, 256, 0, stream>>>(
                in_proj, wt_in, D_MODEL, D_INPROJ, NPAD_WIN);
            k_convert_wt<<<(D_MODEL * D_INNER + 255) / 256, 256, 0, stream>>>(
                out_proj, wt_out, D_INNER, D_MODEL, D_MODEL);

            // zxbcdt = hn @ in_proj   (WMMA bf16, LDS-staged A)
            k_gemm_wmma_bf16<<<grid_in, 256, 0, stream>>>(
                hnb, wt_in, zx, D_INPROJ, D_MODEL, NPAD_ST, 0);

            k_conv_silu<<<(Mtok * D_XBC + 255) / 256, 256, 0, stream>>>(
                zx, conv_w, conv_b, xbc, d);

            k_scan<<<Bn * NHEADS, 256, 0, stream>>>(
                zx, xbc, dt_bias, A_log, Dparam, ybuf, d);

            k_gatenorm<<<Mtok, 256, 0, stream>>>(zx, ybuf, norm_w, ub);

            // residual accumulate: hacc += u @ out_proj   (WMMA bf16)
            k_gemm_wmma_bf16<<<grid_out, 256, 0, stream>>>(
                ub, wt_out, hacc, D_MODEL, D_INNER, D_MODEL, 1);
        }
        float* tmp = hcur; hcur = hacc; hacc = tmp;
    }

    // ---- back end -------------------------------------------------------
    k_zero<<<(Bn * D_MODEL + 255) / 256, 256, 0, stream>>>(pooled, Bn * D_MODEL);
    k_finalln_pool<<<Mtok, 256, 0, stream>>>(hcur, FP(8), FP(9), pooled);
    k_heads<<<Bn, 256, 0, stream>>>(pooled, bio, FP(10), FP(11),
                                    FP(12), FP(13), FP(14), FP(15), (float*)d_out);
    #undef FP
}
